// Plamo2MambaMixer_59648505807445
// MI455X (gfx1250) — matrix operand order repaired and checked by reading, not compiled, not run on hardware
//
#include <hip/hip_runtime.h>

// ---------------------------------------------------------------------------
// Plamo2 Mamba mixer for MI455X (gfx1250, wave32, WMMA + async-LDS).
//   1) gemm_bf16x3  : proj  = hidden @ in_proj_w            [2048 x 4096]
//   2) conv_silu    : hs    = silu(causal depthwise conv)   [2048 x 2048]
//   3) gemm_bf16x3  : ssm_p = hs @ bcdt_w                   [2048 x 256]
//   4) norm_proj    : B,C,ts RMSNorm; dt = ts@dt_proj_w; delta=softplus(dt+b)
//   5) scan         : diagonal linear recurrence; async double-buffered LDS
//   6) gate         : ys = (ys + D*hs) * silu(gate)
//   7) gemm_bf16x3  : out   = ys @ out_proj_w               [2048 x 2048]
// GEMMs: bf16 hi/lo split (3 WMMA / K-step), ping-pong LDS + reg prefetch
// + global_prefetch of tile k+2. Scan: GLOBAL_LOAD_ASYNC_TO_LDS double
// buffering with per-wave balanced ASYNCcnt (11 copies/wave/chunk).
// Workspace requirement: 2048*8592 floats ~= 70.4 MB.
// ---------------------------------------------------------------------------

#define L_TOK 2048
#define H_DIM 2048
#define I_DIM 2048
#define N_ST  64
#define R_DIM 128
#define NH_   16
#define HPH_  128
#define K_CONV 4
#define EPS_  1e-6f

typedef __attribute__((ext_vector_type(16))) __bf16 v16bf;
typedef __attribute__((ext_vector_type(8)))  __bf16 v8bf;
typedef __attribute__((ext_vector_type(4)))  __bf16 v4bf;
typedef __attribute__((ext_vector_type(8)))  float  v8f;
typedef __attribute__((ext_vector_type(4)))  int    v4i;

// address-space-qualified pointee types for the async-LDS builtins
// (signature pattern: (int4 AS1*, int4 AS3*, imm offset, imm cpol))
typedef __attribute__((address_space(1))) v4i g1_v4i;
typedef __attribute__((address_space(3))) v4i l3_v4i;
typedef __attribute__((address_space(1))) int g1_i32;
typedef __attribute__((address_space(3))) int l3_i32;

static __device__ __forceinline__ v16bf cat8(v8bf a, v8bf b) {
  return __builtin_shufflevector(a, b, 0, 1, 2, 3, 4, 5, 6, 7,
                                       8, 9, 10, 11, 12, 13, 14, 15);
}

// --- async global->LDS copy helpers ----------------------------------------
static __device__ __forceinline__ void async_copy_b128(const void* g, void* l) {
#if __has_builtin(__builtin_amdgcn_global_load_async_to_lds_b128)
  __builtin_amdgcn_global_load_async_to_lds_b128((g1_v4i*)g, (l3_v4i*)l, 0, 0);
#else
  unsigned lds = (unsigned)(unsigned long long)l;
  asm volatile("global_load_async_to_lds_b128 %0, %1, off"
               :: "v"(lds), "v"(g) : "memory");
#endif
}

static __device__ __forceinline__ void async_copy_b32(const void* g, void* l) {
#if __has_builtin(__builtin_amdgcn_global_load_async_to_lds_b32)
  __builtin_amdgcn_global_load_async_to_lds_b32((g1_i32*)g, (l3_i32*)l, 0, 0);
#else
  unsigned lds = (unsigned)(unsigned long long)l;
  asm volatile("global_load_async_to_lds_b32 %0, %1, off"
               :: "v"(lds), "v"(g) : "memory");
#endif
}

template <int N>
static __device__ __forceinline__ void wait_async() {
#if __has_builtin(__builtin_amdgcn_s_wait_asynccnt)
  __builtin_amdgcn_s_wait_asynccnt((unsigned short)N);
#else
  asm volatile("s_wait_asynccnt %0" :: "n"(N) : "memory");
#endif
}

// ---------------------------------------------------------------------------
// bf16x3 split GEMM: C[M,N] = A[M,K] * B[K,N], fp32 row-major.
// 128x128 tile, BK=32, 256 threads (8 waves, wave32). Ping-pong LDS buffers:
// tile k+1 fetched to registers and converted/stored while tile k feeds
// 24 WMMAs; one barrier per K-step. Requires M%128==N%128==0, K%32==0.
// ---------------------------------------------------------------------------
#define BM 128
#define BN 128
#define BK 32

__global__ __launch_bounds__(256)
void gemm_bf16x3(const float* __restrict__ Ag, const float* __restrict__ Bg,
                 float* __restrict__ Cg, int M, int N, int K) {
  __shared__ __bf16 Ahi[2][BM][BK];
  __shared__ __bf16 Alo[2][BM][BK];
  __shared__ __bf16 Bthi[2][BN][BK];  // transposed: [n][k]
  __shared__ __bf16 Btlo[2][BN][BK];

  const int tid  = threadIdx.x;
  const int lane = tid & 31;
  const int w    = tid >> 5;
  const int m    = lane & 15;
  const int half = lane >> 4;
  const int blockRow = blockIdx.x * BM;
  const int blockCol = blockIdx.y * BN;

  v8f acc[8] = {};
  float4 av[4];   // A prefetch: 4 groups of 4 consecutive k
  float2 bv[8];   // B prefetch: 2 micro-tiles of 4k x 2n

  auto ldGlobal = [&](int k0) {
#pragma unroll
    for (int i = 0; i < 4; ++i) {
      int idx = tid + i * 256;
      int row = idx >> 3, kg = idx & 7;
      av[i] = *(const float4*)&Ag[(size_t)(blockRow + row) * K + k0 + kg * 4];
    }
#pragma unroll
    for (int i = 0; i < 2; ++i) {
      int idx = tid + i * 256;
      int kg = idx >> 6, np = idx & 63;
#pragma unroll
      for (int j = 0; j < 4; ++j)
        bv[i * 4 + j] = *(const float2*)
            &Bg[(size_t)(k0 + kg * 4 + j) * N + blockCol + np * 2];
    }
  };

  auto stLds = [&](int buf) {
#pragma unroll
    for (int i = 0; i < 4; ++i) {
      int idx = tid + i * 256;
      int row = idx >> 3, kg = idx & 7;
      float4 g = av[i];
      __bf16 h0 = (__bf16)g.x, h1 = (__bf16)g.y;
      __bf16 h2 = (__bf16)g.z, h3 = (__bf16)g.w;
      v4bf hv = {h0, h1, h2, h3};
      v4bf lv = {(__bf16)(g.x - (float)h0), (__bf16)(g.y - (float)h1),
                 (__bf16)(g.z - (float)h2), (__bf16)(g.w - (float)h3)};
      *(v4bf*)&Ahi[buf][row][kg * 4] = hv;
      *(v4bf*)&Alo[buf][row][kg * 4] = lv;
    }
#pragma unroll
    for (int i = 0; i < 2; ++i) {
      int idx = tid + i * 256;
      int kg = idx >> 6, np = idx & 63;
      __bf16 hx[4], hy[4], lx[4], ly[4];
#pragma unroll
      for (int j = 0; j < 4; ++j) {
        float2 g = bv[i * 4 + j];
        hx[j] = (__bf16)g.x; lx[j] = (__bf16)(g.x - (float)hx[j]);
        hy[j] = (__bf16)g.y; ly[j] = (__bf16)(g.y - (float)hy[j]);
      }
      v4bf vx = {hx[0], hx[1], hx[2], hx[3]};
      v4bf vy = {hy[0], hy[1], hy[2], hy[3]};
      v4bf ux = {lx[0], lx[1], lx[2], lx[3]};
      v4bf uy = {ly[0], ly[1], ly[2], ly[3]};
      *(v4bf*)&Bthi[buf][np * 2    ][kg * 4] = vx;
      *(v4bf*)&Bthi[buf][np * 2 + 1][kg * 4] = vy;
      *(v4bf*)&Btlo[buf][np * 2    ][kg * 4] = ux;
      *(v4bf*)&Btlo[buf][np * 2 + 1][kg * 4] = uy;
    }
  };

  ldGlobal(0);
  stLds(0);
  const int nk = K / BK;

  for (int kt = 0; kt < nk; ++kt) {
    __syncthreads();
    const int buf = kt & 1;
    if (kt + 1 < nk) ldGlobal((kt + 1) * BK);
    if (kt + 2 < nk) {
      __builtin_prefetch(
          &Ag[(size_t)(blockRow + (tid >> 3)) * K + (kt + 2) * BK], 0, 1);
      __builtin_prefetch(
          &Bg[(size_t)((kt + 2) * BK + (tid >> 6)) * N + blockCol + (tid & 63) * 2],
          0, 1);
    }

    const __bf16* aph = &Ahi[buf][16 * w + m][half * 8];
    const __bf16* apl = &Alo[buf][16 * w + m][half * 8];
    v16bf ahi = cat8(*(const v8bf*)aph, *(const v8bf*)(aph + 16));
    v16bf alo = cat8(*(const v8bf*)apl, *(const v8bf*)(apl + 16));

#pragma unroll
    for (int s = 0; s < 8; ++s) {
      const __bf16* bph = &Bthi[buf][16 * s + m][half * 8];
      const __bf16* bpl = &Btlo[buf][16 * s + m][half * 8];
      v16bf bhi = cat8(*(const v8bf*)bph, *(const v8bf*)(bph + 16));
      v16bf blo = cat8(*(const v8bf*)bpl, *(const v8bf*)(bpl + 16));
      acc[s] = __builtin_amdgcn_wmma_f32_16x16x32_bf16(
          false, ahi, false, bhi, (short)0, acc[s], false, false);
      acc[s] = __builtin_amdgcn_wmma_f32_16x16x32_bf16(
          false, ahi, false, blo, (short)0, acc[s], false, false);
      acc[s] = __builtin_amdgcn_wmma_f32_16x16x32_bf16(
          false, alo, false, bhi, (short)0, acc[s], false, false);
    }

    if (kt + 1 < nk) stLds(buf ^ 1);
  }

#pragma unroll
  for (int s = 0; s < 8; ++s) {
    int col  = blockCol + 16 * s + m;
    int row0 = blockRow + 16 * w + 8 * half;
#pragma unroll
    for (int r = 0; r < 8; ++r)
      Cg[(size_t)(row0 + r) * N + col] = acc[s][r];
  }
}

// ---------------------------------------------------------------------------
// Causal depthwise conv (K=4, zero history) + SiLU, float4-vectorized.
// ---------------------------------------------------------------------------
__global__ __launch_bounds__(256)
void conv_silu_kernel(const float* __restrict__ proj,
                      const float* __restrict__ cw, float* __restrict__ hs) {
  int idx = blockIdx.x * 256 + threadIdx.x;     // over L*I/4
  int t  = idx >> 9;
  int c4 = (idx & 511) << 2;
  int col = ((c4 >> 7) << 8) + HPH_ + (c4 & (HPH_ - 1));
  float4 wv[4];
#pragma unroll
  for (int j = 0; j < 4; ++j)
    wv[j] = *(const float4*)&cw[(c4 + j) * K_CONV];
  float4 acc = {0.f, 0.f, 0.f, 0.f};
#pragma unroll
  for (int k = 0; k < K_CONV; ++k) {
    int tt = t - (K_CONV - 1) + k;
    if (tt >= 0) {
      float4 v = *(const float4*)&proj[(size_t)tt * 4096 + col];
      acc.x += ((const float*)&wv[0])[k] * v.x;
      acc.y += ((const float*)&wv[1])[k] * v.y;
      acc.z += ((const float*)&wv[2])[k] * v.z;
      acc.w += ((const float*)&wv[3])[k] * v.w;
    }
  }
  float4 o;
  o.x = acc.x * (1.f / (1.f + __expf(-acc.x)));
  o.y = acc.y * (1.f / (1.f + __expf(-acc.y)));
  o.z = acc.z * (1.f / (1.f + __expf(-acc.z)));
  o.w = acc.w * (1.f / (1.f + __expf(-acc.w)));
  *(float4*)&hs[(size_t)idx * 4] = o;
}

// ---------------------------------------------------------------------------
// Per-token RMSNorms + dt projection + softplus.
// ---------------------------------------------------------------------------
__global__ __launch_bounds__(256)
void norm_proj_kernel(const float* __restrict__ ssm,
                      const float* __restrict__ bw, const float* __restrict__ cw,
                      const float* __restrict__ dw, const float* __restrict__ dtw,
                      const float* __restrict__ dtb, float* __restrict__ Bn,
                      float* __restrict__ Cn, float* __restrict__ dl) {
  int t = blockIdx.x, tid = threadIdx.x;
  float x = ssm[(size_t)t * 256 + tid];
  __shared__ float red[256];
  __shared__ float tsn[R_DIM];
  red[tid] = x * x;
  __syncthreads();
  for (int off = 32; off; off >>= 1) {
    if ((tid & 63) < off) red[tid] += red[tid + off];
    __syncthreads();
  }
  if (tid < 64) {
    float r = rsqrtf(red[0] * (1.f / 64.f) + EPS_);
    Bn[(size_t)t * 64 + tid] = x * r * bw[tid];
  } else if (tid < 128) {
    float r = rsqrtf(red[64] * (1.f / 64.f) + EPS_);
    Cn[(size_t)t * 64 + (tid - 64)] = x * r * cw[tid - 64];
  } else {
    float r = rsqrtf((red[128] + red[192]) * (1.f / 128.f) + EPS_);
    tsn[tid - 128] = x * r * dw[tid - 128];
  }
  __syncthreads();
  if (tid < NH_) {
    float acc = 0.f;
#pragma unroll 8
    for (int r = 0; r < R_DIM; ++r) acc += tsn[r] * dtw[r * NH_ + tid];
    float z = acc + dtb[tid];
    dl[(size_t)t * NH_ + tid] = (z > 20.f) ? z : log1pf(__expf(z));
  }
}

// ---------------------------------------------------------------------------
// Selective scan with async double-buffered LDS staging.
// One wave per channel; lane holds states n=lane, lane+32. Workgroup =
// 8 waves = 8 channels (single head). Chunk k+1's B/C/u/delta copies are
// issued with GLOBAL_LOAD_ASYNC_TO_LDS while chunk k's 64 serial t-steps
// run; every wave issues exactly 11 async ops per chunk, so
// s_wait_asynccnt <= 11 == "my previous buffer is complete" (in-order).
// ---------------------------------------------------------------------------
__global__ __launch_bounds__(256)
void scan_kernel(const float* __restrict__ hs, const float* __restrict__ dl,
                 const float* __restrict__ Bn, const float* __restrict__ Cn,
                 const float* __restrict__ A, float* __restrict__ ys) {
  __shared__ float Bs[2][64][N_ST];
  __shared__ float Cs[2][64][N_ST];
  __shared__ float us[2][64][8];
  __shared__ float dsb[2][64];
  const int tid  = threadIdx.x;
  const int w    = tid >> 5;
  const int lane = tid & 31;
  const int cbase = blockIdx.x * 8;
  const int c     = cbase + w;
  const int head  = c >> 7;
  const float a0 = A[(size_t)c * N_ST + lane];
  const float a1 = A[(size_t)c * N_ST + lane + 32];
  float s0 = 0.f, s1 = 0.f;

  // issue one chunk's copies: 4 (B) + 4 (C) + 2 (u) + 1 (delta) per wave
  auto issue_chunk = [&](int buf, int t0) {
#pragma unroll
    for (int i = 0; i < 4; ++i) {
      int idx = tid + i * 256;            // 1024 float4 groups of B and C
      int tt = idx >> 4, ng = idx & 15;
      async_copy_b128(&Bn[(size_t)(t0 + tt) * N_ST + ng * 4],
                      &Bs[buf][tt][ng * 4]);
    }
#pragma unroll
    for (int i = 0; i < 4; ++i) {
      int idx = tid + i * 256;
      int tt = idx >> 4, ng = idx & 15;
      async_copy_b128(&Cn[(size_t)(t0 + tt) * N_ST + ng * 4],
                      &Cs[buf][tt][ng * 4]);
    }
#pragma unroll
    for (int i = 0; i < 2; ++i) {
      int e = w * 64 + i * 32 + lane;     // 512 u elements, 64 per wave
      int tt = e >> 3, cc = e & 7;
      async_copy_b32(&hs[(size_t)(t0 + tt) * I_DIM + cbase + cc],
                     &us[buf][tt][cc]);
    }
    {
      int e = w * 8 + lane;               // 64 delta elements, 8 per wave
      if (lane < 8)
        async_copy_b32(&dl[(size_t)(t0 + e) * NH_ + head], &dsb[buf][e]);
      else
        async_copy_b32(&dl[(size_t)(t0 + (w * 8)) * NH_ + head],
                       &dsb[buf][w * 8]);  // keep per-wave count uniform
    }
  };

  const int NC = L_TOK / 64;
  int buf = 0;
  issue_chunk(0, 0);

  for (int chunk = 0; chunk < NC; ++chunk) {
    const int t0 = chunk * 64;
    if (chunk + 1 < NC) {
      issue_chunk(buf ^ 1, t0 + 64);      // overlap next chunk's DMA
      wait_async<11>();                   // current buffer's 11 are done
    } else {
      wait_async<0>();
    }
    __syncthreads();                      // all waves' copies visible

    for (int tt = 0; tt < 64; ++tt) {
      float d  = dsb[buf][tt];
      float du = d * us[buf][tt][w];
      s0 = s0 * __expf(d * a0) + du * Bs[buf][tt][lane];
      s1 = s1 * __expf(d * a1) + du * Bs[buf][tt][lane + 32];
      float p = s0 * Cs[buf][tt][lane] + s1 * Cs[buf][tt][lane + 32];
#pragma unroll
      for (int off = 16; off; off >>= 1) p += __shfl_xor(p, off, 32);
      if (lane == 0) ys[(size_t)(t0 + tt) * I_DIM + c] = p;
    }
    __syncthreads();                      // buf free for reuse next iter
    buf ^= 1;
  }
}

// ---------------------------------------------------------------------------
// ys = (ys + D*hs) * silu(gate), float4-vectorized.
// ---------------------------------------------------------------------------
__global__ __launch_bounds__(256)
void gate_kernel(const float* __restrict__ proj, const float* __restrict__ hs,
                 const float* __restrict__ D, float* __restrict__ ys) {
  int idx = blockIdx.x * 256 + threadIdx.x;
  int t  = idx >> 9;
  int c4 = (idx & 511) << 2;
  float4 g  = *(const float4*)&proj[(size_t)t * 4096 + ((c4 >> 7) << 8) + (c4 & 127)];
  float4 yv = *(float4*)&ys[(size_t)idx * 4];
  float4 hv = *(const float4*)&hs[(size_t)idx * 4];
  float4 dv = *(const float4*)&D[c4];
  float4 o;
  o.x = (yv.x + dv.x * hv.x) * (g.x * (1.f / (1.f + __expf(-g.x))));
  o.y = (yv.y + dv.y * hv.y) * (g.y * (1.f / (1.f + __expf(-g.y))));
  o.z = (yv.z + dv.z * hv.z) * (g.z * (1.f / (1.f + __expf(-g.z))));
  o.w = (yv.w + dv.w * hv.w) * (g.w * (1.f / (1.f + __expf(-g.w))));
  *(float4*)&ys[(size_t)idx * 4] = o;
}

// ---------------------------------------------------------------------------
extern "C" void kernel_launch(void* const* d_in, const int* in_sizes, int n_in,
                              void* d_out, int out_size, void* d_ws,
                              size_t ws_size, hipStream_t stream) {
  const float* hidden     = (const float*)d_in[0];
  const float* in_proj_w  = (const float*)d_in[1];
  const float* conv_w     = (const float*)d_in[2];
  const float* bcdt_w     = (const float*)d_in[3];
  const float* dt_norm_w  = (const float*)d_in[4];
  const float* B_norm_w   = (const float*)d_in[5];
  const float* C_norm_w   = (const float*)d_in[6];
  const float* dt_proj_w  = (const float*)d_in[7];
  const float* dt_bias    = (const float*)d_in[8];
  const float* A          = (const float*)d_in[9];
  const float* D          = (const float*)d_in[10];
  const float* out_proj_w = (const float*)d_in[11];
  float* out = (float*)d_out;

  float* ws   = (float*)d_ws;
  float* proj = ws;                              // L * 4096
  float* hs   = proj + (size_t)L_TOK * 4096;     // L * 2048
  float* ssm  = hs   + (size_t)L_TOK * I_DIM;    // L * 256
  float* Bn   = ssm  + (size_t)L_TOK * 256;      // L * 64
  float* Cn   = Bn   + (size_t)L_TOK * N_ST;     // L * 64
  float* dl   = Cn   + (size_t)L_TOK * N_ST;     // L * 16
  float* ys   = dl   + (size_t)L_TOK * NH_;      // L * 2048

  gemm_bf16x3<<<dim3(L_TOK / BM, 4096 / BN), 256, 0, stream>>>(
      hidden, in_proj_w, proj, L_TOK, 4096, H_DIM);

  conv_silu_kernel<<<(L_TOK * I_DIM / 4) / 256, 256, 0, stream>>>(
      proj, conv_w, hs);

  gemm_bf16x3<<<dim3(L_TOK / BM, 256 / BN), 256, 0, stream>>>(
      hs, bcdt_w, ssm, L_TOK, 256, I_DIM);

  norm_proj_kernel<<<L_TOK, 256, 0, stream>>>(ssm, B_norm_w, C_norm_w,
                                              dt_norm_w, dt_proj_w, dt_bias,
                                              Bn, Cn, dl);

  scan_kernel<<<I_DIM / 8, 256, 0, stream>>>(hs, dl, Bn, Cn, A, ys);

  gate_kernel<<<(L_TOK * I_DIM / 4) / 256, 256, 0, stream>>>(proj, hs, D, ys);

  gemm_bf16x3<<<dim3(L_TOK / BM, H_DIM / BN), 256, 0, stream>>>(
      ys, out_proj_w, out, L_TOK, H_DIM, I_DIM);
}
